// MultiHeadAttention_38534446580154
// MI455X (gfx1250) — matrix operand order repaired and checked by reading, not compiled, not run on hardware
//
#include <hip/hip_runtime.h>
#include <hip/hip_bf16.h>

typedef __attribute__((ext_vector_type(16))) _Float16 v16h;
typedef __attribute__((ext_vector_type(8)))  _Float16 v8h;
typedef __attribute__((ext_vector_type(4)))  _Float16 v4h;
typedef __attribute__((ext_vector_type(8)))  float    v8f;
typedef __attribute__((ext_vector_type(4)))  float    v4f;
typedef __attribute__((ext_vector_type(4)))  int      v4i;

#define B_  4
#define S_  2048
#define E_  1024
#define H_  16
#define D_  64

#if defined(__AMDGCN__) && __has_builtin(__builtin_amdgcn_global_load_async_to_lds_b128)
#define HAVE_ASYNC_LDS 1
#else
#define HAVE_ASYNC_LDS 0
#endif

typedef __attribute__((address_space(1))) v4i gv4i;   // global int4
typedef __attribute__((address_space(3))) v4i lv4i;   // LDS int4

__device__ __forceinline__ v8f wmma_f16(v16h a, v16h b, v8f c) {
    return __builtin_amdgcn_wmma_f32_16x16x32_f16(
        /*neg_a=*/false, a, /*neg_b=*/false, b,
        /*c_mod=*/(short)0, c, /*reuse_a=*/false, /*reuse_b=*/false);
}

// Copy 32 contiguous f16 (64B) global -> LDS.
__device__ __forceinline__ void copy64B(_Float16* ldst, const _Float16* gsrc) {
#if HAVE_ASYNC_LDS
#pragma unroll
    for (int c = 0; c < 4; ++c)
        __builtin_amdgcn_global_load_async_to_lds_b128(
            (gv4i*)(gsrc + 8 * c), (lv4i*)(ldst + 8 * c), 0, 0);
#else
    *(v16h*)(ldst)      = *(const v16h*)(gsrc);
    *(v16h*)(ldst + 16) = *(const v16h*)(gsrc + 16);
#endif
}

// Copy 16 contiguous f16 (32B) global -> LDS.
__device__ __forceinline__ void copy32B(_Float16* ldst, const _Float16* gsrc) {
#if HAVE_ASYNC_LDS
#pragma unroll
    for (int c = 0; c < 2; ++c)
        __builtin_amdgcn_global_load_async_to_lds_b128(
            (gv4i*)(gsrc + 8 * c), (lv4i*)(ldst + 8 * c), 0, 0);
#else
    *(v16h*)(ldst) = *(const v16h*)(gsrc);
#endif
}

__device__ __forceinline__ void wait_async_copies() {
#if HAVE_ASYNC_LDS
#if __has_builtin(__builtin_amdgcn_s_wait_asynccnt)
    __builtin_amdgcn_s_wait_asynccnt(0);
#else
    asm volatile("s_wait_asynccnt 0" ::: "memory");
#endif
#endif
}

// A-fragment (16x32 f16): lane m=lane%16, half=lane>>4.
__device__ __forceinline__ v16h load_afrag_lds(const _Float16* row, int kbase, int half) {
    return *(const v16h*)(row + kbase + 16 * half);
}

// ---------------- Stage 1a: f32 -> f16 convert (x) ----------------
__global__ __launch_bounds__(256) void cvt_x_kernel(const float* __restrict__ x,
                                                    _Float16* __restrict__ xh) {
    int i = (blockIdx.x * 256 + threadIdx.x) * 4;
    v4f f = *(const v4f*)(x + i);
    v4h o;
    o[0] = (_Float16)f[0]; o[1] = (_Float16)f[1];
    o[2] = (_Float16)f[2]; o[3] = (_Float16)f[3];
    *(v4h*)(xh + i) = o;
}

// ---------------- Stage 1b: transpose+convert W -> wt[n][k] f16 ----------------
__global__ __launch_bounds__(256) void cvt_w_kernel(const float* __restrict__ Wq,
                                                    const float* __restrict__ Wk,
                                                    const float* __restrict__ Wv,
                                                    _Float16* __restrict__ wt) {
    __shared__ float tile[32][33];
    const int mat = blockIdx.z;
    const float* src = (mat == 0) ? Wq : (mat == 1) ? Wk : Wv;
    const int n0 = blockIdx.x * 32, k0 = blockIdx.y * 32;
    const int tx = threadIdx.x, ty = threadIdx.y;
#pragma unroll
    for (int i = 0; i < 4; ++i)
        tile[ty + 8 * i][tx] = src[(size_t)(k0 + ty + 8 * i) * E_ + n0 + tx];
    __syncthreads();
    _Float16* dst = wt + (size_t)mat * E_ * (H_ * D_);
#pragma unroll
    for (int i = 0; i < 4; ++i)
        dst[(size_t)(n0 + ty + 8 * i) * E_ + k0 + tx] = (_Float16)tile[tx][ty + 8 * i];
}

// ---------------- Stage 2: WMMA projection GEMM (double-buffered, async) ----------
// C[m][n] = sum_k xh[m][k] * wt[n][k] + bias[n]
// Tile 128x128, k-depth 64 per LDS stage, 8 waves, wave = 32x64 output.
__global__ __launch_bounds__(256) void proj_gemm_kernel(const _Float16* __restrict__ A,
                                                        const _Float16* __restrict__ Bw,
                                                        const float* __restrict__ bias,
                                                        _Float16* __restrict__ out,
                                                        int vt_mode) {
    __shared__ __align__(32) _Float16 As[2][128][64];   // 16KB x2
    __shared__ __align__(32) _Float16 Bs[2][128][64];   // 16KB x2

    const int tid   = threadIdx.x;
    const int wave  = tid >> 5, lane = tid & 31;
    const int half  = lane >> 4, lanen = lane & 15;
    const int wm    = wave & 3;       // 4 M-blocks of 32 rows
    const int wn    = wave >> 2;      // 2 N-blocks of 64 cols
    const int m0    = blockIdx.y * 128;
    const int n0    = blockIdx.x * 128;
    const int lrow  = tid >> 1;            // 0..127
    const int lcol  = (tid & 1) * 32;      // 0 or 32 halves (64B chunk)

    const _Float16* gA = A  + (size_t)(m0 + lrow) * E_ + lcol;
    const _Float16* gB = Bw + (size_t)(n0 + lrow) * E_ + lcol;

    v8f acc[2][4];
#pragma unroll
    for (int mi = 0; mi < 2; ++mi)
#pragma unroll
        for (int ni = 0; ni < 4; ++ni) acc[mi][ni] = (v8f){};

    const int NT = E_ / 64;   // 16 k-tiles
    // prologue: stage tile 0 into buffer 0
    copy64B(&As[0][lrow][lcol], gA);
    copy64B(&Bs[0][lrow][lcol], gB);

    for (int kt = 0; kt < NT; ++kt) {
        const int cur = kt & 1;
        wait_async_copies();
        __syncthreads();     // publish buf[cur]; license overwrite of buf[cur^1]
        if (kt + 1 < NT) {
            copy64B(&As[cur ^ 1][lrow][lcol], gA + (kt + 1) * 64);
            copy64B(&Bs[cur ^ 1][lrow][lcol], gB + (kt + 1) * 64);
        }
#pragma unroll
        for (int ks = 0; ks < 2; ++ks) {
            v16h af[2], bf[4];
#pragma unroll
            for (int mi = 0; mi < 2; ++mi)
                af[mi] = load_afrag_lds(&As[cur][32 * wm + 16 * mi + lanen][0], 32 * ks, half);
#pragma unroll
            for (int ni = 0; ni < 4; ++ni)
                bf[ni] = *(const v16h*)&Bs[cur][64 * wn + 16 * ni + lanen][32 * ks + 16 * half];
#pragma unroll
            for (int mi = 0; mi < 2; ++mi)
#pragma unroll
                for (int ni = 0; ni < 4; ++ni)
                    acc[mi][ni] = wmma_f16(af[mi], bf[ni], acc[mi][ni]);
        }
    }

#pragma unroll
    for (int mi = 0; mi < 2; ++mi) {
#pragma unroll
        for (int ni = 0; ni < 4; ++ni) {
            const int gn = n0 + 64 * wn + 16 * ni + lanen;
            const float bval = bias[gn];
            const int hh = gn >> 6, dd = gn & 63;
#pragma unroll
            for (int r = 0; r < 8; ++r) {
                const int gm = m0 + 32 * wm + 16 * mi + r + 8 * half;
                const int bb = gm >> 11, ss = gm & 2047;
                const float v = acc[mi][ni][r] + bval;
                size_t idx = vt_mode
                    ? (((size_t)(bb * H_ + hh) * D_ + dd) * S_ + ss)
                    : (((size_t)(bb * H_ + hh) * S_ + ss) * D_ + dd);
                out[idx] = (_Float16)v;
            }
        }
    }
}

// ---------------- Stage 3: flash attention (double-buffered K/V^T tiles) -------
// grid = (B*H, S/128), block = 256 (8 waves). Each wave owns 16 query rows.
__global__ __launch_bounds__(256) void attn_kernel(const _Float16* __restrict__ q,
                                                   const _Float16* __restrict__ kmat,
                                                   const _Float16* __restrict__ vt,
                                                   const int* __restrict__ mask,
                                                   float* __restrict__ out) {
    __shared__ __align__(32) _Float16 Ks[2][64][64];    // [key][d]
    __shared__ __align__(32) _Float16 Vs[2][64][64];    // [d][key]  (V^T tile)
    __shared__ __align__(32) _Float16 Ps[8][16][64];    // per-wave P scratch [q][key]

    const int tid   = threadIdx.x;
    const int wave  = tid >> 5, lane = tid & 31;
    const int half  = lane >> 4, lanen = lane & 15;
    const int bh    = blockIdx.x;
    const int bb    = bh >> 4;
    const int hh    = bh & 15;
    const int q0    = blockIdx.y * 128 + wave * 16;

    // Resident Q fragments (16 rows x 64 d -> 2 k-steps of 32)
    v16h qf[2];
    {
        const _Float16* qrow = q + ((size_t)bh * S_ + q0 + lanen) * D_;
#pragma unroll
        for (int ks = 0; ks < 2; ++ks) {
            qf[ks] = *(const v16h*)(qrow + 32 * ks + 16 * half);
        }
    }

    float mrun[8], lrun[8];
    v8f Oacc[4];
#pragma unroll
    for (int r = 0; r < 8; ++r) { mrun[r] = -3.0e38f; lrun[r] = 0.f; }
#pragma unroll
    for (int dt = 0; dt < 4; ++dt) Oacc[dt] = (v8f){};

    const int ldr = tid >> 2;          // 0..63
    const int ldc = (tid & 3) * 16;    // 32B chunk
    const _Float16* gK = kmat + ((size_t)bh * S_ + ldr) * D_ + ldc;
    const _Float16* gV = vt + ((size_t)bh * D_ + ldr) * S_ + ldc;
    const int* gM = mask + bb * S_;

    const int NT = S_ / 64;   // 32 key tiles
    copy32B(&Ks[0][ldr][ldc], gK);
    copy32B(&Vs[0][ldr][ldc], gV);

    for (int it = 0; it < NT; ++it) {
        const int t0 = it * 64;
        const int cur = it & 1;
        wait_async_copies();
        __syncthreads();     // publish buf[cur]; license overwrite of buf[cur^1]
        if (it + 1 < NT) {
            copy32B(&Ks[cur ^ 1][ldr][ldc], gK + (size_t)(t0 + 64) * D_);
            copy32B(&Vs[cur ^ 1][ldr][ldc], gV + (t0 + 64));
        }

        // scores: S = Q * K^T, scaled + masked
        float sc[4][8];
#pragma unroll
        for (int nt = 0; nt < 4; ++nt) {
            v8f s = (v8f){};
#pragma unroll
            for (int ks = 0; ks < 2; ++ks) {
                v16h bf = *(const v16h*)&Ks[cur][16 * nt + lanen][32 * ks + 16 * half];
                s = wmma_f16(qf[ks], bf, s);
            }
            const float madd = (gM[t0 + 16 * nt + lanen] == 0) ? -1.0e30f : 0.f;
#pragma unroll
            for (int r = 0; r < 8; ++r) sc[nt][r] = s[r] * 0.125f + madd;
        }

        // online softmax: row max across nt tiles + 16-lane group
        float tm[8];
#pragma unroll
        for (int r = 0; r < 8; ++r)
            tm[r] = fmaxf(fmaxf(sc[0][r], sc[1][r]), fmaxf(sc[2][r], sc[3][r]));
#pragma unroll
        for (int off = 8; off >= 1; off >>= 1)
#pragma unroll
            for (int r = 0; r < 8; ++r)
                tm[r] = fmaxf(tm[r], __shfl_xor(tm[r], off, 32));

        float corr[8];
#pragma unroll
        for (int r = 0; r < 8; ++r) {
            const float mnew = fmaxf(mrun[r], tm[r]);
            corr[r] = __expf(mrun[r] - mnew);
            mrun[r] = mnew;
        }

        float rs[8];
#pragma unroll
        for (int r = 0; r < 8; ++r) rs[r] = 0.f;
#pragma unroll
        for (int nt = 0; nt < 4; ++nt)
#pragma unroll
            for (int r = 0; r < 8; ++r) {
                const float p = __expf(sc[nt][r] - mrun[r]);
                rs[r] += p;
                Ps[wave][r + 8 * half][16 * nt + lanen] = (_Float16)p;
            }
#pragma unroll
        for (int off = 8; off >= 1; off >>= 1)
#pragma unroll
            for (int r = 0; r < 8; ++r) rs[r] += __shfl_xor(rs[r], off, 32);
#pragma unroll
        for (int r = 0; r < 8; ++r) lrun[r] = lrun[r] * corr[r] + rs[r];
#pragma unroll
        for (int dt = 0; dt < 4; ++dt)
#pragma unroll
            for (int r = 0; r < 8; ++r) Oacc[dt][r] *= corr[r];

        // make this wave's Ps stores visible to its own ds loads (in-order per wave)
        asm volatile("s_wait_dscnt 0" ::: "memory");

        // O += P * V  (A = Ps[q][t], B = Vs[d][t])
#pragma unroll
        for (int dt = 0; dt < 4; ++dt) {
#pragma unroll
            for (int ks = 0; ks < 2; ++ks) {
                v16h af = *(const v16h*)&Ps[wave][lanen][32 * ks + 16 * half];
                v16h bf = *(const v16h*)&Vs[cur][16 * dt + lanen][32 * ks + 16 * half];
                Oacc[dt] = wmma_f16(af, bf, Oacc[dt]);
            }
        }
    }

    // epilogue: divide by l, write y[b][s][h*64+d]
#pragma unroll
    for (int dt = 0; dt < 4; ++dt) {
#pragma unroll
        for (int r = 0; r < 8; ++r) {
            const int srow = q0 + r + 8 * half;
            const float l = lrun[r];
            const float val = (l > 0.f) ? Oacc[dt][r] / l : 0.f;
            out[((size_t)bb * S_ + srow) * (H_ * D_) + hh * D_ + 16 * dt + lanen] = val;
        }
    }
}

extern "C" void kernel_launch(void* const* d_in, const int* in_sizes, int n_in,
                              void* d_out, int out_size, void* d_ws, size_t ws_size,
                              hipStream_t stream) {
    const float* x    = (const float*)d_in[0];
    const int*   mask = (const int*)d_in[1];
    const float* Wq   = (const float*)d_in[2];
    const float* bq   = (const float*)d_in[3];
    const float* Wk   = (const float*)d_in[4];
    const float* bk   = (const float*)d_in[5];
    const float* Wv   = (const float*)d_in[6];
    const float* bv   = (const float*)d_in[7];
    float* out = (float*)d_out;

    char* ws = (char*)d_ws;
    const size_t xh_bytes = (size_t)B_ * S_ * E_ * 2;          // 16 MB
    const size_t wt_bytes = (size_t)3 * E_ * H_ * D_ * 2;      //  6 MB
    const size_t qk_bytes = (size_t)B_ * S_ * H_ * D_ * 2;     // 16 MB each
    _Float16* xh  = (_Float16*)ws;
    _Float16* wt  = (_Float16*)(ws + xh_bytes);
    _Float16* qb  = (_Float16*)(ws + xh_bytes + wt_bytes);
    _Float16* kb  = (_Float16*)(ws + xh_bytes + wt_bytes + qk_bytes);
    _Float16* vtb = (_Float16*)(ws + xh_bytes + wt_bytes + 2 * qk_bytes);

    // Stage 1: conversions
    cvt_x_kernel<<<(B_ * S_ * E_) / (256 * 4), 256, 0, stream>>>(x, xh);
    cvt_w_kernel<<<dim3(32, 32, 3), dim3(32, 8), 0, stream>>>(Wq, Wk, Wv, wt);

    // Stage 2: QKV projections (WMMA GEMMs)
    const size_t wstep = (size_t)E_ * H_ * D_;
    proj_gemm_kernel<<<dim3(8, 64), 256, 0, stream>>>(xh, wt,             bq, qb,  0);
    proj_gemm_kernel<<<dim3(8, 64), 256, 0, stream>>>(xh, wt + wstep,     bk, kb,  0);
    proj_gemm_kernel<<<dim3(8, 64), 256, 0, stream>>>(xh, wt + 2 * wstep, bv, vtb, 1);

    // Stage 3: flash attention (WMMA)
    attn_kernel<<<dim3(B_ * H_, S_ / 128), 256, 0, stream>>>(qb, kb, vtb, mask, out);
}
